// GNN_10746008174936
// MI455X (gfx1250) — matrix operand order repaired and checked by reading, not compiled, not run on hardware
//
#include <hip/hip_runtime.h>
#include <stdint.h>

// ---- problem constants (match reference) ----
static constexpr int Nn  = 50000;
static constexpr int Ne  = 800000;
static constexpr int Et  = Ne + Nn;      // edges incl. self loops

typedef __attribute__((ext_vector_type(16))) __bf16        v16bf;
typedef __attribute__((ext_vector_type(8)))  float         v8f;
typedef __attribute__((ext_vector_type(4)))  unsigned int  u32x4;

union ABFrag { v16bf v; u32x4 u[2]; };

__device__ __forceinline__ unsigned short f2bf(float f) {
  unsigned u = __float_as_uint(f);
  u += 0x7FFFu + ((u >> 16) & 1u);           // round-to-nearest-even
  return (unsigned short)(u >> 16);
}

// order-preserving float <-> uint map for atomicMax-based segment max
__device__ __forceinline__ unsigned fmap(float x) {
  unsigned u = __float_as_uint(x);
  return (u & 0x80000000u) ? ~u : (u | 0x80000000u);
}
__device__ __forceinline__ float funmap(unsigned u) {
  return __uint_as_float((u & 0x80000000u) ? (u & 0x7FFFFFFFu) : ~u);
}

__global__ void k_f32_to_bf16(const float* __restrict__ in, unsigned short* __restrict__ out, int n) {
  int i = blockIdx.x * blockDim.x + threadIdx.x;
  if (i < n) out[i] = f2bf(in[i]);
}

__global__ void k_relu_to_bf16(const float* __restrict__ in, unsigned short* __restrict__ out, int n) {
  int i = blockIdx.x * blockDim.x + threadIdx.x;
  if (i < n) { float v = in[i]; out[i] = f2bf(v > 0.f ? v : 0.f); }
}

// Bt[col][k] = bf16( col<nh ? Wl[k][col] : Wr[k][col-nh] ) ; Wl/Wr are K x nh row-major
__global__ void k_prep_bt(const float* __restrict__ Wl, const float* __restrict__ Wr,
                          int K, int nh, unsigned short* __restrict__ Bt) {
  int t = blockIdx.x * blockDim.x + threadIdx.x;
  if (t >= 2 * nh * K) return;
  int col = t / K, k = t - col * K;
  float v = (col < nh) ? Wl[(size_t)k * nh + col] : Wr[(size_t)k * nh + (col - nh)];
  Bt[(size_t)col * K + k] = f2bf(v);
}

__global__ void k_bias_init(float* __restrict__ out, const float* __restrict__ b, int n, int cmask) {
  int i = blockIdx.x * blockDim.x + threadIdx.x;
  if (i < n) out[i] = b[i & cmask];      // cols is a power of two
}

__global__ void k_zero_u32(unsigned* __restrict__ p, int n) {
  int i = blockIdx.x * blockDim.x + threadIdx.x;
  if (i < n) p[i] = 0u;
}

// Y(M x ncols) = A(M x KK, bf16 row-major) @ Bt^T (Bt is ncols x KK, bf16) ; fp32 accum.
// 8 waves/block, NT 16-col tiles per wave, one 16-row M tile per block.
// Software-pipelined: k-step fragments double-buffered in registers so loads
// for step ks+1 are in flight while the WMMA chain for step ks executes.
template <int NT, int KK>
__global__ __launch_bounds__(256)
void k_gemm_wmma(const unsigned short* __restrict__ A, const unsigned short* __restrict__ Bt,
                 float* __restrict__ Y, int ncols) {
  constexpr int NK = KK / 32;
  const int wave  = threadIdx.x >> 5;
  const int lane  = threadIdx.x & 31;
  const int mrow  = blockIdx.x * 16 + (lane & 15);
  const int ahalf = (lane >> 4) << 3;   // A frag: K in {0..7,16..23} / {8..15,24..31}
  const int bhalf = (lane >> 4) << 3;   // B frag: K in {0..7,16..23} / {8..15,24..31}

  v8f acc[NT] = {};
  const unsigned short* arow = A + (size_t)mrow * KK + ahalf;
  const unsigned short* brow[NT];
#pragma unroll
  for (int t = 0; t < NT; ++t) {
    const int col = (wave * NT + t) * 16 + (lane & 15);
    brow[t] = Bt + (size_t)col * KK + bhalf;
  }

  ABFrag a[2];
  ABFrag b[2][NT];
  // prologue: k-step 0
  a[0].u[0] = *(const u32x4*)(arow);
  a[0].u[1] = *(const u32x4*)(arow + 16);
#pragma unroll
  for (int t = 0; t < NT; ++t) {
    b[0][t].u[0] = *(const u32x4*)(brow[t]);
    b[0][t].u[1] = *(const u32x4*)(brow[t] + 16);
  }
#pragma unroll
  for (int ks = 0; ks < NK; ++ks) {
    const int cur = ks & 1, nxt = cur ^ 1;
    if (ks + 1 < NK) {                     // prefetch next k-step
      const int k0 = (ks + 1) * 32;
      a[nxt].u[0] = *(const u32x4*)(arow + k0);
      a[nxt].u[1] = *(const u32x4*)(arow + k0 + 16);
#pragma unroll
      for (int t = 0; t < NT; ++t) {
        b[nxt][t].u[0] = *(const u32x4*)(brow[t] + k0);
        b[nxt][t].u[1] = *(const u32x4*)(brow[t] + k0 + 16);
      }
    }
#pragma unroll
    for (int t = 0; t < NT; ++t)
      acc[t] = __builtin_amdgcn_wmma_f32_16x16x32_bf16(
          false, a[cur].v, false, b[cur][t].v, (short)0, acc[t], false, false);
  }

  const int rbase = blockIdx.x * 16 + ((lane >> 4) << 3);
#pragma unroll
  for (int t = 0; t < NT; ++t) {
    const int col = (wave * NT + t) * 16 + (lane & 15);
#pragma unroll
    for (int r = 0; r < 8; ++r)
      Y[(size_t)(rbase + r) * ncols + col] = acc[t][r];
  }
}

// wave per edge: logit[e][h] = sum_c att[h][c]*leaky(xl[src][h,c]+xr[dst][h,c]); seg-max via atomicMax
template <int H, int C>
__global__ void k_edge_logits(const float* __restrict__ Y, const int* __restrict__ ei,
                              const float* __restrict__ att,
                              float* __restrict__ logit, unsigned* __restrict__ nmax) {
  constexpr int HC = H * C;
  constexpr int CH = HC / 32;     // values per lane
  constexpr int GL = 32 / H;      // lanes per head group
  const int e = (int)((blockIdx.x * (unsigned)blockDim.x + threadIdx.x) >> 5);
  if (e >= Et) return;
  const int lane = threadIdx.x & 31;
  int src, dst;
  if (e < Ne) { src = ei[e]; dst = ei[Ne + e]; } else { src = dst = e - Ne; }
  const float* xl = Y + (size_t)src * (2 * HC);
  const float* xr = Y + (size_t)dst * (2 * HC) + HC;
  const int j0 = lane * CH;
  const int h  = j0 / C;
  float s = 0.f;
#pragma unroll
  for (int i = 0; i < CH; ++i) {
    int j = j0 + i;
    float v = xl[j] + xr[j];
    v = v > 0.f ? v : 0.2f * v;   // leaky_relu(0.2)
    s += att[j] * v;              // att is H*C contiguous
  }
#pragma unroll
  for (int m = 1; m < GL; m <<= 1) s += __shfl_xor(s, m, 32);
  if ((lane & (GL - 1)) == 0) {
    logit[(size_t)e * H + h] = s;
    atomicMax(nmax + (size_t)dst * H + h, fmap(s));
  }
}

template <int H>
__global__ void k_edge_exp(const int* __restrict__ ei, float* __restrict__ logit,
                           const unsigned* __restrict__ nmax, float* __restrict__ denom) {
  const int t = blockIdx.x * blockDim.x + threadIdx.x;
  if (t >= Et * H) return;
  const int e = t / H, h = t - e * H;
  const int dst = (e < Ne) ? ei[Ne + e] : (e - Ne);
  const float ex = __expf(logit[t] - funmap(nmax[(size_t)dst * H + h]));
  logit[t] = ex;
  atomicAdd(denom + (size_t)dst * H + h, ex);
}

template <int H, int C>
__global__ void k_edge_scatter(const float* __restrict__ Y, const int* __restrict__ ei,
                               const float* __restrict__ exw, const float* __restrict__ denom,
                               float* __restrict__ out) {
  constexpr int HC = H * C;
  constexpr int CH = HC / 32;
  const int e = (int)((blockIdx.x * (unsigned)blockDim.x + threadIdx.x) >> 5);
  if (e >= Et) return;
  const int lane = threadIdx.x & 31;
  int src, dst;
  if (e < Ne) { src = ei[e]; dst = ei[Ne + e]; } else { src = dst = e - Ne; }
  const int j0 = lane * CH;
  const int h  = j0 / C;
  const float alpha = exw[(size_t)e * H + h] / denom[(size_t)dst * H + h];
  const float* xl = Y + (size_t)src * (2 * HC) + j0;
  float* o = out + (size_t)dst * HC + j0;
#pragma unroll
  for (int i = 0; i < CH; ++i) atomicAdd(o + i, alpha * xl[i]);
}

extern "C" void kernel_launch(void* const* d_in, const int* in_sizes, int n_in,
                              void* d_out, int out_size, void* d_ws, size_t ws_size,
                              hipStream_t stream) {
  (void)in_sizes; (void)n_in; (void)out_size; (void)ws_size;
  const float* x    = (const float*)d_in[0];
  const int*   ei   = (const int*)d_in[1];   // (2,E) int32: [0..E)=src, [E..2E)=dst
  const float* Wl1  = (const float*)d_in[3];
  const float* Wr1  = (const float*)d_in[4];
  const float* att1 = (const float*)d_in[5];
  const float* b1   = (const float*)d_in[6];
  const float* Wl2  = (const float*)d_in[7];
  const float* Wr2  = (const float*)d_in[8];
  const float* att2 = (const float*)d_in[9];
  const float* b2   = (const float*)d_in[10];

  char* ws = (char*)d_ws;
  size_t off = 0;
  auto take = [&](size_t bytes) {
    char* p = ws + off; off = (off + bytes + 255) & ~(size_t)255; return p;
  };
  unsigned short* xb   = (unsigned short*)take((size_t)Nn * 256 * 2);  // x bf16; reused as h bf16
  unsigned short* bt1  = (unsigned short*)take((size_t)512 * 256 * 2); // [Wl1|Wr1]^T bf16
  unsigned short* bt2  = (unsigned short*)take((size_t)256 * 256 * 2); // [Wl2|Wr2]^T bf16
  float*          Y1   = (float*)take((size_t)Nn * 512 * 4);           // [xl1|xr1]; reused as Y2
  float*          out1 = (float*)take((size_t)Nn * 256 * 4);
  float*          lg1  = (float*)take((size_t)Et * 4 * 4);             // reused layer2
  unsigned*       mx1  = (unsigned*)take((size_t)Nn * 4 * 4);          // reused layer2
  float*          dn1  = (float*)take((size_t)Nn * 4 * 4);             // reused layer2
  unsigned short* hb  = xb;
  float*    Y2  = Y1;
  float*    lg2 = lg1;
  unsigned* mx2 = mx1;
  float*    dn2 = dn1;
  float*    out2 = (float*)d_out;

  const int B = 256;
  auto cdiv = [](long long a, long long b) { return (int)((a + b - 1) / b); };

  // ---------------- layer 1 ----------------
  k_f32_to_bf16<<<cdiv((long long)Nn * 256, B), B, 0, stream>>>(x, xb, Nn * 256);
  k_prep_bt<<<cdiv(512 * 256, B), B, 0, stream>>>(Wl1, Wr1, 256, 256, bt1);
  k_prep_bt<<<cdiv(256 * 256, B), B, 0, stream>>>(Wl2, Wr2, 256, 128, bt2);
  k_gemm_wmma<4, 256><<<Nn / 16, 256, 0, stream>>>(xb, bt1, Y1, 512);
  k_bias_init<<<cdiv((long long)Nn * 256, B), B, 0, stream>>>(out1, b1, Nn * 256, 255);
  k_zero_u32<<<cdiv(Nn * 4, B), B, 0, stream>>>(mx1, Nn * 4);
  k_zero_u32<<<cdiv(Nn * 4, B), B, 0, stream>>>((unsigned*)dn1, Nn * 4);
  k_edge_logits<4, 64><<<Et / 8, B, 0, stream>>>(Y1, ei, att1, lg1, mx1);
  k_edge_exp<4><<<cdiv((long long)Et * 4, B), B, 0, stream>>>(ei, lg1, mx1, dn1);
  k_edge_scatter<4, 64><<<Et / 8, B, 0, stream>>>(Y1, ei, lg1, dn1, out1);
  k_relu_to_bf16<<<cdiv((long long)Nn * 256, B), B, 0, stream>>>(out1, hb, Nn * 256);
  // ---------------- layer 2 ----------------
  k_gemm_wmma<2, 256><<<Nn / 16, 256, 0, stream>>>(hb, bt2, Y2, 256);
  k_bias_init<<<cdiv((long long)Nn * 128, B), B, 0, stream>>>(out2, b2, Nn * 128, 127);
  k_zero_u32<<<cdiv(Nn, B), B, 0, stream>>>(mx2, Nn);
  k_zero_u32<<<cdiv(Nn, B), B, 0, stream>>>((unsigned*)dn2, Nn);
  k_edge_logits<1, 128><<<Et / 8, B, 0, stream>>>(Y2, ei, att2, lg2, mx2);
  k_edge_exp<1><<<cdiv(Et, B), B, 0, stream>>>(ei, lg2, mx2, dn2);
  k_edge_scatter<1, 128><<<Et / 8, B, 0, stream>>>(Y2, ei, lg2, dn2, out2);
}
